// NeuralResidualVectorQuantizer_1108101562603
// MI455X (gfx1250) — matrix operand order repaired and checked by reading, not compiled, not run on hardware
//
#include <hip/hip_runtime.h>
#include <hip/hip_bf16.h>

// ---- WMMA types (CDNA5 gfx1250, wave32) ----
typedef __attribute__((ext_vector_type(16))) __bf16 v16bf;
typedef __attribute__((ext_vector_type(8)))  float  v8f;

union FragBF { v16bf v; unsigned int u[8]; };

// Problem constants (from reference setup_inputs)
#define RVQ_B    8
#define RVQ_C    256
#define RVQ_T    2048
#define RVQ_NQ   8
#define RVQ_BINS 1024
#define RVQ_N    (RVQ_B * RVQ_T)            // 16384 rows of x_flat
#define QSTE_ELEMS ((size_t)RVQ_B * RVQ_C * RVQ_T)  // 4194304
#define IDX_ELEMS  ((size_t)RVQ_N * RVQ_NQ)         // 131072

__device__ __forceinline__ unsigned short f32_to_bf16_rne(float f) {
    unsigned int u = __float_as_uint(f);
    unsigned int r = u + 0x7FFFu + ((u >> 16) & 1u);
    return (unsigned short)(r >> 16);
}
__device__ __forceinline__ float bf16u_to_f32(unsigned short h) {
    return __uint_as_float(((unsigned int)h) << 16);
}

// ------------------------------------------------------------------
// Kernel 1: codebook -> bf16 hi/lo split + c2[q][bin]; zero loss slot
// grid = NQ*BINS blocks of 256 threads (one thread per channel c)
// ------------------------------------------------------------------
__global__ __launch_bounds__(256) void rvq_prep_kernel(
    const float* __restrict__ cb,
    unsigned short* __restrict__ cb_hi,
    unsigned short* __restrict__ cb_lo,
    float* __restrict__ c2,
    float* __restrict__ loss_slot)
{
    __shared__ float red[8];
    const int g = blockIdx.x;            // q*BINS + bin
    const int c = threadIdx.x;           // channel
    const size_t off = (size_t)g * RVQ_C + c;
    float v = cb[off];
    unsigned short hi = f32_to_bf16_rne(v);
    unsigned short lo = f32_to_bf16_rne(v - bf16u_to_f32(hi));
    cb_hi[off] = hi;
    cb_lo[off] = lo;

    float s = v * v;
    #pragma unroll
    for (int o = 16; o; o >>= 1) s += __shfl_xor(s, o, 32);
    if ((c & 31) == 0) red[c >> 5] = s;
    __syncthreads();
    if (c < 8) {
        float t = red[c];
        #pragma unroll
        for (int o = 4; o; o >>= 1) t += __shfl_xor(t, o, 32);
        if (c == 0) c2[g] = t;
    }
    if (g == 0 && c == 0) *loss_slot = 0.0f;   // zeroed every call (deterministic)
}

// ------------------------------------------------------------------
// Kernel 2: WMMA distance + argmin, 2 M-tiles per wave.
// grid = N/32 blocks, 256 threads (8 waves). Block owns a 32-row x tile
// (two 16-row WMMA tiles) converted to bf16 hi/lo in LDS; wave w handles
// quantizer q = w, scanning all 1024 bins in 16-wide tiles.
// Per B hi/lo fragment pair: 6 WMMAs on two independent accumulator
// chains (halves codebook L2 traffic, hides WMMA RAW latency).
// score(n,bin) = c2[bin] - 2*dot(x_n, c_bin)   (x^2 dropped: row-constant)
// ------------------------------------------------------------------
__global__ __launch_bounds__(256) void rvq_argmin_kernel(
    const float* __restrict__ x,
    const unsigned short* __restrict__ cb_hi,
    const unsigned short* __restrict__ cb_lo,
    const float* __restrict__ c2,
    int* __restrict__ idx_i,
    float* __restrict__ idx_f)
{
    __shared__ unsigned short As_hi[32 * RVQ_C];
    __shared__ unsigned short As_lo[32 * RVQ_C];

    const int n0 = blockIdx.x * 32;           // first x_flat row of tile
    const int b  = n0 / RVQ_T;                // tile never crosses a batch (T%32==0)
    const int t0 = n0 % RVQ_T;
    const int tid = threadIdx.x;

    // --- load + convert x tile: x_flat[n][c] = x[b][c][t0 + r], t contiguous ---
    {
        const int r  = tid & 31;
        const int c0 = tid >> 5;
        const float* xb = x + (size_t)b * RVQ_C * RVQ_T + t0 + r;
        for (int c = c0; c < RVQ_C; c += 8) {
            float f = xb[(size_t)c * RVQ_T];
            unsigned short hi = f32_to_bf16_rne(f);
            unsigned short lo = f32_to_bf16_rne(f - bf16u_to_f32(hi));
            As_hi[r * RVQ_C + c] = hi;
            As_lo[r * RVQ_C + c] = lo;
        }
    }
    __syncthreads();

    const int q    = tid >> 5;     // wave id == quantizer index
    const int lane = tid & 31;
    const int nn   = lane & 15;    // N (bin within 16-tile) / M (row) index
    const int hl   = lane >> 4;    // lane half

    const unsigned short* cbh = cb_hi + (size_t)q * RVQ_BINS * RVQ_C;
    const unsigned short* cbl = cb_lo + (size_t)q * RVQ_BINS * RVQ_C;
    const float* c2q = c2 + q * RVQ_BINS;

    // Preload A_hi fragments for both M-tiles, all 8 K-chunks, into registers.
    // A 16x32 bf16 layout: lanes0-15: M=lane, K in {0..7}∪{16..23};
    //                      lanes16-31: M=lane-16, K in {8..15}∪{24..31}.
    FragBF Ah0[8], Ah1[8];
    #pragma unroll
    for (int kc = 0; kc < 8; ++kc) {
        const int base0 = nn * RVQ_C + kc * 32 + hl * 8;
        const int base1 = (16 + nn) * RVQ_C + kc * 32 + hl * 8;
        #pragma unroll
        for (int v = 0; v < 4; ++v) {
            Ah0[kc].u[v]     = *(const unsigned int*)&As_hi[base0 + 2 * v];
            Ah0[kc].u[4 + v] = *(const unsigned int*)&As_hi[base0 + 16 + 2 * v];
            Ah1[kc].u[v]     = *(const unsigned int*)&As_hi[base1 + 2 * v];
            Ah1[kc].u[4 + v] = *(const unsigned int*)&As_hi[base1 + 16 + 2 * v];
        }
    }

    float minv0[8], minv1[8];
    int   mini0[8], mini1[8];
    #pragma unroll
    for (int v = 0; v < 8; ++v) {
        minv0[v] = 3.402823466e38f; mini0[v] = 0;
        minv1[v] = 3.402823466e38f; mini1[v] = 0;
    }

    for (int j = 0; j < RVQ_BINS / 16; ++j) {
        const int bin = j * 16 + nn;
        v8f acc0 = {0.f, 0.f, 0.f, 0.f, 0.f, 0.f, 0.f, 0.f};
        v8f acc1 = {0.f, 0.f, 0.f, 0.f, 0.f, 0.f, 0.f, 0.f};

        #pragma unroll
        for (int kc = 0; kc < 8; ++kc) {
            // A_lo fragments from LDS (compiler may hoist if registers allow)
            FragBF Al0, Al1;
            const int abase0 = nn * RVQ_C + kc * 32 + hl * 8;
            const int abase1 = (16 + nn) * RVQ_C + kc * 32 + hl * 8;
            #pragma unroll
            for (int v = 0; v < 4; ++v) {
                Al0.u[v]     = *(const unsigned int*)&As_lo[abase0 + 2 * v];
                Al0.u[4 + v] = *(const unsigned int*)&As_lo[abase0 + 16 + 2 * v];
                Al1.u[v]     = *(const unsigned int*)&As_lo[abase1 + 2 * v];
                Al1.u[4 + v] = *(const unsigned int*)&As_lo[abase1 + 16 + 2 * v];
            }

            // B 32x16 bf16 layout: lane L -> N = L&15; same K order as A:
            // lanes0-15 K in {0..7}∪{16..23}, lanes16-31 K in {8..15}∪{24..31}
            FragBF Bh, Bl;
            const size_t bbase = (size_t)bin * RVQ_C + kc * 32 + hl * 8;
            #pragma unroll
            for (int v = 0; v < 4; ++v) {
                Bh.u[v]     = *(const unsigned int*)&cbh[bbase + 2 * v];
                Bh.u[4 + v] = *(const unsigned int*)&cbh[bbase + 16 + 2 * v];
                Bl.u[v]     = *(const unsigned int*)&cbl[bbase + 2 * v];
                Bl.u[4 + v] = *(const unsigned int*)&cbl[bbase + 16 + 2 * v];
            }

            // Two independent accumulator chains: 6 WMMAs per B fragment pair
            acc0 = __builtin_amdgcn_wmma_f32_16x16x32_bf16(
                       false, Ah0[kc].v, false, Bh.v, (short)0, acc0, false, false);
            acc1 = __builtin_amdgcn_wmma_f32_16x16x32_bf16(
                       false, Ah1[kc].v, false, Bh.v, (short)0, acc1, false, false);
            acc0 = __builtin_amdgcn_wmma_f32_16x16x32_bf16(
                       false, Ah0[kc].v, false, Bl.v, (short)0, acc0, false, false);
            acc1 = __builtin_amdgcn_wmma_f32_16x16x32_bf16(
                       false, Ah1[kc].v, false, Bl.v, (short)0, acc1, false, false);
            acc0 = __builtin_amdgcn_wmma_f32_16x16x32_bf16(
                       false, Al0.v,     false, Bh.v, (short)0, acc0, false, false);
            acc1 = __builtin_amdgcn_wmma_f32_16x16x32_bf16(
                       false, Al1.v,     false, Bh.v, (short)0, acc1, false, false);
        }

        // C/D layout: VGPR v -> row (v + 8*hl), column nn
        const float c2v = c2q[bin];
        #pragma unroll
        for (int v = 0; v < 8; ++v) {
            float d0 = __builtin_fmaf(-2.0f, acc0[v], c2v);
            if (d0 < minv0[v] || (d0 == minv0[v] && bin < mini0[v])) {
                minv0[v] = d0; mini0[v] = bin;
            }
            float d1 = __builtin_fmaf(-2.0f, acc1[v], c2v);
            if (d1 < minv1[v] || (d1 == minv1[v] && bin < mini1[v])) {
                minv1[v] = d1; mini1[v] = bin;
            }
        }
    }

    // Argmin across the 16 lanes of each half (columns), first-index tie-break.
    #pragma unroll
    for (int v = 0; v < 8; ++v) {
        float d0 = minv0[v]; int i0 = mini0[v];
        float d1 = minv1[v]; int i1 = mini1[v];
        #pragma unroll
        for (int o = 8; o; o >>= 1) {           // xor<16 keeps halves separate
            float od0 = __shfl_xor(d0, o, 32);
            int   oi0 = __shfl_xor(i0, o, 32);
            if (od0 < d0 || (od0 == d0 && oi0 < i0)) { d0 = od0; i0 = oi0; }
            float od1 = __shfl_xor(d1, o, 32);
            int   oi1 = __shfl_xor(i1, o, 32);
            if (od1 < d1 || (od1 == d1 && oi1 < i1)) { d1 = od1; i1 = oi1; }
        }
        if (nn == 0) {                           // lane 0 -> rows 0..7, lane 16 -> rows 8..15
            const int row = v + 8 * hl;
            const int na  = n0 + row;            // tile 0
            const int nb  = n0 + 16 + row;       // tile 1
            idx_i[na * RVQ_NQ + q] = i0;
            idx_f[na * RVQ_NQ + q] = (float)i0;  // d_out is float-typed
            idx_i[nb * RVQ_NQ + q] = i1;
            idx_f[nb * RVQ_NQ + q] = (float)i1;
        }
    }
}

// ------------------------------------------------------------------
// Kernel 3: gather winners (fp32 codebook), sum over q, write q_ste in
// (B,C,T) layout (t-contiguous stores), accumulate loss = 2*mean(diff^2).
// grid = B*C blocks of 256 threads; thread covers 8 t-values.
// ------------------------------------------------------------------
__global__ __launch_bounds__(256) void rvq_gather_kernel(
    const float* __restrict__ x,
    const float* __restrict__ cb,
    const int* __restrict__ idx_i,
    float* __restrict__ q_ste,
    float* __restrict__ loss_slot)
{
    __shared__ float red[8];
    const int bc = blockIdx.x;          // b*C + c
    const int b  = bc >> 8;
    const int c  = bc & 255;
    const float* xp = x     + (size_t)bc * RVQ_T;
    float*       op = q_ste + (size_t)bc * RVQ_T;

    float acc = 0.0f;
    #pragma unroll
    for (int k = 0; k < RVQ_T / 256; ++k) {
        const int t = threadIdx.x + k * 256;
        const int n = b * RVQ_T + t;
        float s = 0.0f;
        #pragma unroll
        for (int q = 0; q < RVQ_NQ; ++q) {
            const int id = idx_i[n * RVQ_NQ + q];
            s += cb[((size_t)q * RVQ_BINS + id) * RVQ_C + c];
        }
        const float xv = xp[t];
        op[t] = xv + (s - xv);          // q_ste, same float ops as reference
        const float d = s - xv;
        acc += d * d;
    }

    #pragma unroll
    for (int o = 16; o; o >>= 1) acc += __shfl_xor(acc, o, 32);
    if ((threadIdx.x & 31) == 0) red[threadIdx.x >> 5] = acc;
    __syncthreads();
    if (threadIdx.x < 8) {
        float t2 = red[threadIdx.x];
        #pragma unroll
        for (int o = 4; o; o >>= 1) t2 += __shfl_xor(t2, o, 32);
        if (threadIdx.x == 0)
            atomicAdd(loss_slot, t2 * (2.0f / (float)QSTE_ELEMS));
    }
}

// ------------------------------------------------------------------
extern "C" void kernel_launch(void* const* d_in, const int* in_sizes, int n_in,
                              void* d_out, int out_size, void* d_ws, size_t ws_size,
                              hipStream_t stream) {
    (void)in_sizes; (void)n_in; (void)out_size; (void)ws_size;

    const float* x  = (const float*)d_in[0];          // (8,256,2048) f32
    const float* cb = (const float*)d_in[1];          // (8,1024,256) f32
    // d_in[2] sample_rate, d_in[3] bandwidth: unused

    float* out    = (float*)d_out;
    float* q_ste  = out;                              // 4194304 floats
    float* idx_f  = out + QSTE_ELEMS;                 // 131072 floats
    float* loss   = out + QSTE_ELEMS + IDX_ELEMS;     // 1 float

    // Workspace layout (~8.6 MB):
    char* ws = (char*)d_ws;
    unsigned short* cb_hi = (unsigned short*)ws;                                  // 4 MB
    unsigned short* cb_lo = (unsigned short*)(ws + (size_t)4 * 1024 * 1024);      // 4 MB
    float*          c2    = (float*)(ws + (size_t)8 * 1024 * 1024);               // 32 KB
    int*            idx_i = (int*)  (ws + (size_t)8 * 1024 * 1024 + 64 * 1024);   // 512 KB

    rvq_prep_kernel  <<<RVQ_NQ * RVQ_BINS, 256, 0, stream>>>(cb, cb_hi, cb_lo, c2, loss);
    rvq_argmin_kernel<<<RVQ_N / 32,        256, 0, stream>>>(x, cb_hi, cb_lo, c2, idx_i, idx_f);
    rvq_gather_kernel<<<RVQ_B * RVQ_C,     256, 0, stream>>>(x, cb, idx_i, q_ste, loss);
}
